// ESN_79542794322004
// MI455X (gfx1250) — matrix operand order repaired and checked by reading, not compiled, not run on hardware
//
#include <hip/hip_runtime.h>
#include <hip/hip_bf16.h>
#include <math.h>

#define T_STEPS 512
#define BATCH   32
#define FI      64
#define FO      64
#define NRES    2048
#define LEAK    0.3f

typedef __attribute__((ext_vector_type(16))) __bf16 v16bf;
typedef __attribute__((ext_vector_type(8)))  __bf16 v8bf;
typedef __attribute__((ext_vector_type(8)))  float  v8f;
typedef __attribute__((ext_vector_type(2)))  float  v2f;

// ---------------- fp32 -> bf16 conversion (grid-stride) ----------------
__global__ __launch_bounds__(256) void f32_to_bf16_kernel(const float* __restrict__ src,
                                                          __bf16* __restrict__ dst, int n) {
    int i = blockIdx.x * blockDim.x + threadIdx.x;
    int stride = gridDim.x * blockDim.x;
    for (; i < n; i += stride) dst[i] = (__bf16)src[i];
}

// ---------------- one ESN timestep ----------------
// pre[b][n] = sum_k xb_in[b][k]*Wb[n][k] + sum_f Ub[t][b][f]*Winb[n][f] + bias[n]
// x_next    = clip(0.7*x_old + 0.3*tanh(pre)); write fp32 to X[t], bf16 to xb_out.
// 256 waves: wave w -> (mt = w&1, nt = w>>1); tile rows b = mt*16..+15, cols n = nt*16..+15.
__global__ __launch_bounds__(256) void esn_step_kernel(
    const __bf16* __restrict__ Wb,     // [NRES][NRES] bf16 (row n = column of x@W^T)
    const __bf16* __restrict__ Winb,   // [NRES][FI]   bf16
    const __bf16* __restrict__ Ub,     // [T][B][FI]   bf16
    const __bf16* __restrict__ xb_in,  // [B][NRES]    bf16 (state at t, unused when t==0)
    __bf16*       __restrict__ xb_out, // [B][NRES]    bf16 (state at t+1)
    const float*  __restrict__ bias,   // [NRES]
    float*        __restrict__ X,      // [T][B][NRES] fp32 output region
    int t)
{
    const int tid  = threadIdx.x;
    const int lane = tid & 31;
    const int w    = blockIdx.x * (blockDim.x >> 5) + (tid >> 5);
    const int mt   = w & 1;
    const int nt   = w >> 1;
    const int n0   = nt * 16;
    const int lo   = lane & 15;
    const int hi   = lane >> 4;
    const int bRow = mt * 16 + lo;   // A-matrix row for this lane
    const int nCol = n0 + lo;        // B/C/D column for this lane

    v8f c = {0.f, 0.f, 0.f, 0.f, 0.f, 0.f, 0.f, 0.f};

    // ---- input projection: K = 64 over (U[t], Win) ----
    {
        const __bf16* urow = Ub + ((size_t)t * BATCH + bRow) * FI;
        const __bf16* wrow = Winb + (size_t)nCol * FI;
#pragma unroll
        for (int kk = 0; kk < FI; kk += 32) {
            v16bf a = *(const v16bf*)(urow + kk + hi * 16);
            v16bf b = *(const v16bf*)(wrow + kk + hi * 16);
            c = __builtin_amdgcn_wmma_f32_16x16x32_bf16(false, a, false, b, (short)0, c, false, false);
        }
    }

    // ---- recurrent part: K = 2048 over (x_t, W) ; x_0 == 0 so skip at t==0 ----
    if (t > 0) {
        const __bf16* xrow = xb_in + (size_t)bRow * NRES;
        const __bf16* wrow = Wb + (size_t)nCol * NRES;
#pragma unroll 4
        for (int k = 0; k < NRES; k += 32) {
            v16bf a = *(const v16bf*)(xrow + k + hi * 16);
            v16bf b = *(const v16bf*)(wrow + k + hi * 16);
            c = __builtin_amdgcn_wmma_f32_16x16x32_bf16(false, a, false, b, (short)0, c, false, false);
        }
    }

    // ---- fused elementwise: leak + tanh + clip; write fp32 X[t] and bf16 state ----
    const float  bn    = bias[nCol];
    const float* xprev = X + (size_t)(t - 1) * (BATCH * NRES);  // only read when t>0
    float*       xcur  = X + (size_t)t * (BATCH * NRES);
#pragma unroll
    for (int r = 0; r < 8; ++r) {
        const int m = hi * 8 + r;          // C/D row within 16x16 tile
        const int b = mt * 16 + m;
        const size_t idx = (size_t)b * NRES + nCol;
        float pre  = c[r] + bn;
        float xold = (t > 0) ? xprev[idx] : 0.0f;
        float xn   = (1.0f - LEAK) * xold + LEAK * tanhf(pre);
        xn = fminf(1.0f, fmaxf(-1.0f, xn));
        xcur[idx]   = xn;
        xb_out[idx] = (__bf16)xn;
    }
}

// ---------------- readout: Y[T*B, FO] = X[T*B, NRES] @ Wout[NRES, FO] ----------------
// fp32 WMMA 16x16x4 directly on fp32 X (no conversion pass over 128 MB).
__global__ __launch_bounds__(256) void esn_out_kernel(
    const float* __restrict__ X,     // [T*B][NRES]
    const float* __restrict__ Wout,  // [NRES][FO]
    float*       __restrict__ Y)     // [T*B][FO]
{
    const int tid  = threadIdx.x;
    const int lane = tid & 31;
    const int w    = blockIdx.x * (blockDim.x >> 5) + (tid >> 5);
    const int ntile = w & 3;           // FO/16 = 4 column tiles
    const int mtile = w >> 2;          // (T*B)/16 = 1024 row tiles
    const int m0   = mtile * 16;
    const int n0   = ntile * 16;
    const int lo   = lane & 15;
    const int hi   = lane >> 4;

    const float* arow = X + (size_t)(m0 + lo) * NRES;
    v8f c = {0.f, 0.f, 0.f, 0.f, 0.f, 0.f, 0.f, 0.f};

#pragma unroll 4
    for (int k = 0; k < NRES; k += 4) {
        v2f a = *(const v2f*)(arow + k + hi * 2);   // A 16x4 f32: lane row, K = hi*2 + {0,1}
        v2f b;
        b[0] = Wout[(size_t)(k + hi * 2 + 0) * FO + n0 + lo];  // B 4x16 f32: col in lanes
        b[1] = Wout[(size_t)(k + hi * 2 + 1) * FO + n0 + lo];
        c = __builtin_amdgcn_wmma_f32_16x16x4_f32(false, a, false, b, (short)0, c, false, false);
    }

#pragma unroll
    for (int r = 0; r < 8; ++r) {
        const int m = m0 + hi * 8 + r;
        Y[(size_t)m * FO + n0 + lo] = c[r];
    }
}

extern "C" void kernel_launch(void* const* d_in, const int* in_sizes, int n_in,
                              void* d_out, int out_size, void* d_ws, size_t ws_size,
                              hipStream_t stream) {
    const float* U    = (const float*)d_in[0];  // [T,B,FI]
    const float* W    = (const float*)d_in[1];  // [N,N]
    const float* Win  = (const float*)d_in[2];  // [N,FI]
    const float* bias = (const float*)d_in[3];  // [N]
    const float* Wout = (const float*)d_in[4];  // [N,FO]

    float* X = (float*)d_out;                                   // [T,B,N]
    float* Y = X + (size_t)T_STEPS * BATCH * NRES;              // [T,B,FO]

    // workspace layout (bf16 copies + ping-pong state)
    __bf16* Wb    = (__bf16*)d_ws;                              // N*N      = 4,194,304
    __bf16* Winb  = Wb   + (size_t)NRES * NRES;                 // N*FI     =   131,072
    __bf16* Ub    = Winb + (size_t)NRES * FI;                   // T*B*FI   = 1,048,576
    __bf16* xbuf0 = Ub   + (size_t)T_STEPS * BATCH * FI;        // B*N      =    65,536
    __bf16* xbuf1 = xbuf0 + (size_t)BATCH * NRES;               // B*N

    // 1) convert static operands to bf16 (idempotent, re-done every call)
    f32_to_bf16_kernel<<<1024, 256, 0, stream>>>(W, Wb, NRES * NRES);
    f32_to_bf16_kernel<<<64,   256, 0, stream>>>(Win, Winb, NRES * FI);
    f32_to_bf16_kernel<<<512,  256, 0, stream>>>(U, Ub, T_STEPS * BATCH * FI);

    // 2) sequential scan: 512 stream-ordered step kernels (graph-capturable)
    //    256 tiles -> 256 waves -> 32 blocks of 256 threads (8 waves)
    for (int t = 0; t < T_STEPS; ++t) {
        __bf16* xin  = (t & 1) ? xbuf0 : xbuf1;   // read  (ignored at t==0)
        __bf16* xout = (t & 1) ? xbuf1 : xbuf0;   // write
        esn_step_kernel<<<32, 256, 0, stream>>>(Wb, Winb, Ub, xin, xout, bias, X, t);
    }

    // 3) readout GEMM: 4096 tiles -> 512 blocks of 8 waves
    esn_out_kernel<<<512, 256, 0, stream>>>(X, Wout, Y);
}
